// SF_RCP_63763084476704
// MI455X (gfx1250) — matrix operand order repaired and checked by reading, not compiled, not run on hardware
//
#include <hip/hip_runtime.h>

#define BATCH 4
#define NPTS  8192
#define S1P   4096
#define S2P   2048
#define KNN   32
#define CHUNK 256

typedef __attribute__((ext_vector_type(16))) _Float16 v16h;
typedef __attribute__((ext_vector_type(8)))  float    v8f;

// ---------------------------------------------------------------------------
// gfx1250 async global->LDS path (guarded: falls back to sync staging if the
// toolchain doesn't declare the builtin).
#if defined(__AMDGCN__) && __has_builtin(__builtin_amdgcn_global_load_async_to_lds_b32)
#define ASYNC_LDS 1
#else
#define ASYNC_LDS 0
#endif

#if ASYNC_LDS
__device__ __forceinline__ void async_copy_b32(const float* g, float* l) {
    __builtin_amdgcn_global_load_async_to_lds_b32(
        (__attribute__((address_space(1))) int*)(void*)g,
        (__attribute__((address_space(3))) int*)(void*)l,
        0, 0);
}
template <int N>
__device__ __forceinline__ void async_wait() {
#if __has_builtin(__builtin_amdgcn_s_wait_asynccnt)
    __builtin_amdgcn_s_wait_asynccnt(N);
#else
    asm volatile("s_wait_asynccnt %0" :: "i"(N) : "memory");
#endif
}
#endif

// ---------------------------------------------------------------------------
// [B,3,N] -> [B,N,3]
__global__ void k_transpose(const float* __restrict__ src, float* __restrict__ dst, int n) {
    int t = blockIdx.x * blockDim.x + threadIdx.x;
    int total = BATCH * n * 3;
    if (t >= total) return;
    int c = t % 3;
    int p = (t / 3) % n;
    int b = t / (3 * n);
    dst[t] = src[(size_t)b * 3 * n + (size_t)c * n + p];
}

// ---------------------------------------------------------------------------
// Farthest point sampling, fully LDS-resident: the point cloud (<=96KB) and the
// running min-dist array (<=32KB) live in LDS so the 4096-round serial argmax
// chain runs on ds_load/VALU latency instead of L2 round trips. One block
// (1024 thr) per batch; tie-break prefers lowest index (jnp.argmax semantics).
__global__ void k_fps(const float* __restrict__ xyz, int* __restrict__ idx_out, int n, int s) {
    const int T = 1024;
    int b = blockIdx.x;
    int tid = threadIdx.x;
    const float* px = xyz + (size_t)b * n * 3;
    __shared__ float spt[NPTS * 3];   // 96 KB
    __shared__ float sd[NPTS];        // 32 KB
    __shared__ float rv[1024];
    __shared__ int   ri[1024];
    __shared__ int   s_far;
    for (int j = tid; j < n * 3; j += T) spt[j] = px[j];
    for (int j = tid; j < n; j += T) sd[j] = 1e10f;
    if (tid == 0) s_far = 0;
    __syncthreads();
    for (int i = 0; i < s; i++) {
        int far = s_far;
        if (tid == 0) idx_out[b * s + i] = far;
        float cx = spt[far * 3 + 0], cy = spt[far * 3 + 1], cz = spt[far * 3 + 2];
        float bv = -1.0f;
        int   bi = 0x7fffffff;
        for (int j = tid; j < n; j += T) {
            float dx = spt[j * 3 + 0] - cx;
            float dy = spt[j * 3 + 1] - cy;
            float dz = spt[j * 3 + 2] - cz;
            float d  = dx * dx + dy * dy + dz * dz;
            float od = sd[j];
            float nd = d < od ? d : od;
            sd[j] = nd;
            if (nd > bv) { bv = nd; bi = j; }
        }
        rv[tid] = bv; ri[tid] = bi;
        __syncthreads();
        for (int off = T / 2; off > 0; off >>= 1) {
            if (tid < off) {
                float v2 = rv[tid + off]; int i2 = ri[tid + off];
                if (v2 > rv[tid] || (v2 == rv[tid] && i2 < ri[tid])) {
                    rv[tid] = v2; ri[tid] = i2;
                }
            }
            __syncthreads();
        }
        if (tid == 0) s_far = ri[0];
        __syncthreads();
    }
}

// ---------------------------------------------------------------------------
// Gather sampled xyz: dst[b,s,:] = src[b, idx[b,s], :]
__global__ void k_gather3(const float* __restrict__ src, const int* __restrict__ idx,
                          int n, int s, float* __restrict__ dst) {
    int t = blockIdx.x * blockDim.x + threadIdx.x;
    int total = BATCH * s;
    if (t >= total) return;
    int b = t / s;
    int id = idx[t];
    const float* p = src + ((size_t)b * n + id) * 3;
    dst[(size_t)t * 3 + 0] = p[0];
    dst[(size_t)t * 3 + 1] = p[1];
    dst[(size_t)t * 3 + 2] = p[2];
}

// ---------------------------------------------------------------------------
// kNN: one thread per query. Candidate tiles double-buffered through LDS with
// gfx1250 async loads (ASYNCcnt) when available: issue chunk c+1 into the other
// buffer, wait asynccnt<=3 (3 in-flight loads/thread, in-order completion =>
// chunk c landed), barrier, process. Sorted insertion keeps ascending distance,
// ties keep lower index (top_k order).
__global__ void k_knn(const float* __restrict__ qxyz, const float* __restrict__ cand,
                      int n, int s, int* __restrict__ knn_out) {
    int q = blockIdx.x * 256 + threadIdx.x;   // s % 256 == 0 -> one batch per block
    int b = q / s;
    __shared__ float cp[2][CHUNK * 3];
    const float* pcand = cand + (size_t)b * n * 3;
    float qx = qxyz[(size_t)q * 3 + 0];
    float qy = qxyz[(size_t)q * 3 + 1];
    float qz = qxyz[(size_t)q * 3 + 2];
    float bd[KNN]; int bi[KNN];
    #pragma unroll
    for (int t = 0; t < KNN; t++) { bd[t] = 3.4e38f; bi[t] = 0; }
#if ASYNC_LDS
    for (int w = threadIdx.x; w < CHUNK * 3; w += 256)          // prime chunk 0
        async_copy_b32(pcand + w, &cp[0][w]);
#endif
    for (int base = 0; base < n; base += CHUNK) {
        int cur = (base / CHUNK) & 1;
#if ASYNC_LDS
        if (base + CHUNK < n) {
            const float* src = pcand + (size_t)(base + CHUNK) * 3;
            for (int w = threadIdx.x; w < CHUNK * 3; w += 256)  // prefetch next chunk
                async_copy_b32(src + w, &cp[1 - cur][w]);
            async_wait<3>();                                    // current chunk landed
        } else {
            async_wait<0>();
        }
        __syncthreads();
#else
        for (int w = threadIdx.x; w < CHUNK * 3; w += 256)
            cp[cur][w] = pcand[(size_t)base * 3 + w];
        __builtin_prefetch(pcand + (size_t)(base + CHUNK) * 3, 0, 3);
        __syncthreads();
#endif
        for (int c = 0; c < CHUNK; c++) {
            float dx = cp[cur][c * 3 + 0] - qx;
            float dy = cp[cur][c * 3 + 1] - qy;
            float dz = cp[cur][c * 3 + 2] - qz;
            float d = dx * dx + dy * dy + dz * dz;
            if (d < bd[KNN - 1]) {
                int pos = KNN - 1;
                while (pos > 0 && bd[pos - 1] > d) {
                    bd[pos] = bd[pos - 1]; bi[pos] = bi[pos - 1]; pos--;
                }
                bd[pos] = d; bi[pos] = base + c;
            }
        }
        __syncthreads();
    }
    #pragma unroll
    for (int t = 0; t < KNN; t++) knn_out[(size_t)q * KNN + t] = bi[t];
}

// ---------------------------------------------------------------------------
// Grouping with K-padded rows: X0[b,s,k,:] = [cand_xyz[idx]-new_xyz | feat[idx] | 0...]
// Row stride CINP (WMMA K multiple) so the GEMM A-fragment loads are aligned float4s.
template <int CINP>
__global__ void k_group(const float* __restrict__ cand_xyz, const float* __restrict__ newxyz,
                        const int* __restrict__ knn, const float* __restrict__ feat,
                        int featC, int n, int s, float* __restrict__ X0) {
    int t = blockIdx.x * blockDim.x + threadIdx.x;
    int total = BATCH * s * KNN;
    if (t >= total) return;
    int q = t >> 5;            // (b*s + si)
    int b = q / s;
    int id = knn[t];
    const float* p  = cand_xyz + ((size_t)b * n + id) * 3;
    const float* nx = newxyz + (size_t)q * 3;
    float* o = X0 + (size_t)t * CINP;
    o[0] = p[0] - nx[0];
    o[1] = p[1] - nx[1];
    o[2] = p[2] - nx[2];
    const float* f = feat + ((size_t)b * n + id) * featC;
    int c = 0;
    for (; c < featC; c++) o[3 + c] = f[c];
    for (c = 3 + featC; c < CINP; c++) o[c] = 0.0f;
}

// ---------------------------------------------------------------------------
// 1x1-conv GEMM on tensor cores: Y[R,COUT] = X[R,CINP(pad)] * W[COUT,CIN]^T + bias.
// One wave per 16-row M-tile; v_wmma_f32_16x16x32_f16, f32 accumulate.
// A fragment: per lane two runs of 8 contiguous floats (K = 32ks + 16g + 8run + h),
// loaded as unconditional aligned float4 pairs (X rows are CINP-padded with zeros).
// B fragment: W staged once per block into LDS as f16, zero-padded to CINP; per lane
// 16 contiguous halves (kb = 32ks + 16g + h) -> one aligned v16h LDS read.
template <int CINP, int CIN, int COUT>
__global__ void k_gemm_wmma(const float* __restrict__ X, const float* __restrict__ W,
                            const float* __restrict__ bias, float* __restrict__ Y, int R) {
    constexpr int KSTEPS = CINP / 32;
    constexpr int NTILES = COUT / 16;
    __shared__ _Float16 lw[COUT * CINP];
    __shared__ float    lb[COUT];
    int tid = threadIdx.x;
    // stage W (f32 -> f16) + bias into LDS; clamped-index load + select (no branches)
    for (int i = tid; i < COUT * CINP; i += 256) {
        int r = i / CINP, c = i % CINP;
        int cc = c < CIN ? c : 0;
        float v = W[r * CIN + cc];
        lw[i] = (_Float16)(c < CIN ? v : 0.0f);
    }
    for (int i = tid; i < COUT; i += 256) lb[i] = bias[i];
    __syncthreads();

    int wid  = (blockIdx.x * 256 + tid) >> 5;
    int lane = tid & 31;
    int mtiles = R >> 4;
    if (wid >= mtiles) return;           // wave-uniform (EXEC stays all-ones)
    int g   = lane >> 4;
    int l16 = lane & 15;
    int row0 = wid * 16;
    const float* xr = X + (size_t)(row0 + l16) * CINP;

    v16h afr[KSTEPS];
    #pragma unroll
    for (int ks = 0; ks < KSTEPS; ks++) {
        #pragma unroll
        for (int run = 0; run < 2; run++) {
            const float4* xp = (const float4*)(xr + 32 * ks + 16 * g + 8 * run);
            float4 f0 = xp[0];
            float4 f1 = xp[1];
            int hb = run * 8;
            afr[ks][hb + 0] = (_Float16)f0.x;
            afr[ks][hb + 1] = (_Float16)f0.y;
            afr[ks][hb + 2] = (_Float16)f0.z;
            afr[ks][hb + 3] = (_Float16)f0.w;
            afr[ks][hb + 4] = (_Float16)f1.x;
            afr[ks][hb + 5] = (_Float16)f1.y;
            afr[ks][hb + 6] = (_Float16)f1.z;
            afr[ks][hb + 7] = (_Float16)f1.w;
        }
    }
    #pragma unroll
    for (int nt = 0; nt < NTILES; nt++) {
        int ncol = nt * 16 + l16;
        v8f acc = {0.f, 0.f, 0.f, 0.f, 0.f, 0.f, 0.f, 0.f};
        #pragma unroll
        for (int ks = 0; ks < KSTEPS; ks++) {
            v16h bm = *(const v16h*)(lw + ncol * CINP + 32 * ks + 16 * g);
            acc = __builtin_amdgcn_wmma_f32_16x16x32_f16(
                false, afr[ks], false, bm, (short)0, acc, false, false);
        }
        float bz = lb[ncol];
        float* yr = Y + (size_t)row0 * COUT + (size_t)(8 * g) * COUT + ncol;
        #pragma unroll
        for (int j = 0; j < 8; j++) {
            yr[(size_t)j * COUT] = acc[j] + bz;
        }
    }
}

// ---------------------------------------------------------------------------
__global__ void k_zero(float* p, int n) {
    int t = blockIdx.x * blockDim.x + threadIdx.x;
    if (t < n) p[t] = 0.0f;
}

// Per-(batch,channel) sum / sumsq for instance norm. LDS partials, one global
// atomic per (b,c) per block.
__global__ void k_stats(const float* __restrict__ Y, int rowsPerB, int C,
                        float* __restrict__ sum, float* __restrict__ sumsq) {
    __shared__ float ls[BATCH * 64];
    __shared__ float lq[BATCH * 64];
    int tid = threadIdx.x;
    int nbc = BATCH * C;
    for (int i = tid; i < nbc; i += blockDim.x) { ls[i] = 0.0f; lq[i] = 0.0f; }
    __syncthreads();
    size_t total = (size_t)BATCH * rowsPerB * C;
    size_t stride = (size_t)gridDim.x * blockDim.x;
    for (size_t t = (size_t)blockIdx.x * blockDim.x + tid; t < total; t += stride) {
        float v = Y[t];
        int c = (int)(t % C);
        int b = (int)(t / ((size_t)rowsPerB * C));
        atomicAdd(&ls[b * C + c], v);
        atomicAdd(&lq[b * C + c], v * v);
    }
    __syncthreads();
    for (int i = tid; i < nbc; i += blockDim.x) {
        atomicAdd(&sum[i], ls[i]);
        atomicAdd(&sumsq[i], lq[i]);
    }
}

// In-place instance norm (affine=False) + ReLU.
__global__ void k_norm_relu(float* __restrict__ Y, const float* __restrict__ sum,
                            const float* __restrict__ sumsq, int rowsPerB, int C) {
    size_t t = (size_t)blockIdx.x * blockDim.x + threadIdx.x;
    size_t total = (size_t)BATCH * rowsPerB * C;
    if (t >= total) return;
    int c = (int)(t % C);
    int b = (int)(t / ((size_t)rowsPerB * C));
    float cnt = (float)rowsPerB;
    float m = sum[b * C + c] / cnt;
    float v = sumsq[b * C + c] / cnt - m * m;
    float x = (Y[t] - m) * rsqrtf(v + 1e-5f);
    Y[t] = x > 0.0f ? x : 0.0f;
}

// Max over K neighbors; optional [B,C,S] transposed output.
__global__ void k_maxpool(const float* __restrict__ Y, int s, int C,
                          float* __restrict__ out, int transposed) {
    int t = blockIdx.x * blockDim.x + threadIdx.x;
    int total = BATCH * s * C;
    if (t >= total) return;
    int c  = t % C;
    int si = (t / C) % s;
    int b  = t / (C * s);
    const float* base = Y + (((size_t)b * s + si) * KNN) * C + c;
    float m = -3.4e38f;
    #pragma unroll 4
    for (int k = 0; k < KNN; k++) {
        float v = base[(size_t)k * C];
        m = v > m ? v : m;
    }
    if (transposed) out[(size_t)b * C * s + (size_t)c * s + si] = m;
    else            out[((size_t)b * s + si) * C + c] = m;
}

// [B,s,3] -> [B,3,s]
__global__ void k_out_pc(const float* __restrict__ nx, int s, float* __restrict__ out) {
    int t = blockIdx.x * blockDim.x + threadIdx.x;
    int total = BATCH * s * 3;
    if (t >= total) return;
    int c  = t % 3;
    int si = (t / 3) % s;
    int b  = t / (3 * s);
    out[(size_t)b * 3 * s + (size_t)c * s + si] = nx[t];
}

__global__ void k_copy_i32(const int* __restrict__ src, int* __restrict__ dst, int n) {
    int t = blockIdx.x * blockDim.x + threadIdx.x;
    if (t < n) dst[t] = src[t];
}

// ---------------------------------------------------------------------------
extern "C" void kernel_launch(void* const* d_in, const int* in_sizes, int n_in,
                              void* d_out, int out_size, void* d_ws, size_t ws_size,
                              hipStream_t stream) {
    (void)in_sizes; (void)n_in; (void)out_size; (void)ws_size;
    const float* pc      = (const float*)d_in[0];
    const float* feature = (const float*)d_in[1];
    const float* W1[3] = {(const float*)d_in[2], (const float*)d_in[4], (const float*)d_in[6]};
    const float* B1[3] = {(const float*)d_in[3], (const float*)d_in[5], (const float*)d_in[7]};
    const float* W2[3] = {(const float*)d_in[8], (const float*)d_in[10], (const float*)d_in[12]};
    const float* B2[3] = {(const float*)d_in[9], (const float*)d_in[11], (const float*)d_in[13]};

    // workspace layout (float units)
    float* ws = (float*)d_ws;
    size_t o = 0;
    float* xyz   = ws + o; o += (size_t)BATCH * NPTS * 3;
    float* feats = ws + o; o += (size_t)BATCH * NPTS * 3;
    int*   fps1  = (int*)(ws + o); o += (size_t)BATCH * S1P;
    int*   fps2  = (int*)(ws + o); o += (size_t)BATCH * S2P;
    float* nx1   = ws + o; o += (size_t)BATCH * S1P * 3;
    float* nx2   = ws + o; o += (size_t)BATCH * S2P * 3;
    int*   knn1  = (int*)(ws + o); o += (size_t)BATCH * S1P * KNN;
    int*   knn2  = (int*)(ws + o); o += (size_t)BATCH * S2P * KNN;
    float* bufA  = ws + o; o += (size_t)16777216;   // max R*C = B*S2*K*64 = B*S1*K*32
    float* bufB  = ws + o; o += (size_t)16777216;
    float* pool1 = ws + o; o += (size_t)BATCH * S1P * 32;
    float* ssum  = ws + o; o += 256;
    float* ssq   = ws + o; o += 256;

    float* out = (float*)d_out;
    const size_t PC1_OFF  = 0;
    const size_t PC2_OFF  = (size_t)BATCH * 3 * S1P;                 // 49152
    const size_t FEAT_OFF = PC2_OFF + (size_t)BATCH * 3 * S2P;       // 73728
    const size_t FPS1_OFF = FEAT_OFF + (size_t)BATCH * 64 * S2P;     // 598016
    const size_t FPS2_OFF = FPS1_OFF + (size_t)BATCH * S1P;          // 614400

    const int R1 = BATCH * S1P * KNN;   // 524288
    const int R2 = BATCH * S2P * KNN;   // 262144

    // [B,3,N] -> [B,N,3]
    k_transpose<<<(BATCH * NPTS * 3 + 255) / 256, 256, 0, stream>>>(pc, xyz, NPTS);
    k_transpose<<<(BATCH * NPTS * 3 + 255) / 256, 256, 0, stream>>>(feature, feats, NPTS);

    // ---- SA1 --------------------------------------------------------------
    k_fps<<<BATCH, 1024, 0, stream>>>(xyz, fps1, NPTS, S1P);
    k_gather3<<<(BATCH * S1P + 255) / 256, 256, 0, stream>>>(xyz, fps1, NPTS, S1P, nx1);
    k_knn<<<BATCH * S1P / 256, 256, 0, stream>>>(nx1, xyz, NPTS, S1P, knn1);
    k_group<32><<<(R1 + 255) / 256, 256, 0, stream>>>(xyz, nx1, knn1, feats, 3, NPTS, S1P, bufA);

    // layer 0: A(pad32,Cin6) -> B
    k_zero<<<2, 256, 0, stream>>>(ssum, 512);
    k_gemm_wmma<32, 6, 32><<<(R1 / 16) / 8, 256, 0, stream>>>(bufA, W1[0], B1[0], bufB, R1);
    k_stats<<<256, 256, 0, stream>>>(bufB, S1P * KNN, 32, ssum, ssq);
    k_norm_relu<<<(int)(((size_t)R1 * 32 + 255) / 256), 256, 0, stream>>>(bufB, ssum, ssq, S1P * KNN, 32);
    // layer 1: B -> A
    k_zero<<<2, 256, 0, stream>>>(ssum, 512);
    k_gemm_wmma<32, 32, 32><<<(R1 / 16) / 8, 256, 0, stream>>>(bufB, W1[1], B1[1], bufA, R1);
    k_stats<<<256, 256, 0, stream>>>(bufA, S1P * KNN, 32, ssum, ssq);
    k_norm_relu<<<(int)(((size_t)R1 * 32 + 255) / 256), 256, 0, stream>>>(bufA, ssum, ssq, S1P * KNN, 32);
    // layer 2: A -> B
    k_zero<<<2, 256, 0, stream>>>(ssum, 512);
    k_gemm_wmma<32, 32, 32><<<(R1 / 16) / 8, 256, 0, stream>>>(bufA, W1[2], B1[2], bufB, R1);
    k_stats<<<256, 256, 0, stream>>>(bufB, S1P * KNN, 32, ssum, ssq);
    k_norm_relu<<<(int)(((size_t)R1 * 32 + 255) / 256), 256, 0, stream>>>(bufB, ssum, ssq, S1P * KNN, 32);

    k_maxpool<<<(BATCH * S1P * 32 + 255) / 256, 256, 0, stream>>>(bufB, S1P, 32, pool1, 0);

    // ---- SA2 --------------------------------------------------------------
    k_fps<<<BATCH, 1024, 0, stream>>>(nx1, fps2, S1P, S2P);
    k_gather3<<<(BATCH * S2P + 255) / 256, 256, 0, stream>>>(nx1, fps2, S1P, S2P, nx2);
    k_knn<<<BATCH * S2P / 256, 256, 0, stream>>>(nx2, nx1, S1P, S2P, knn2);
    k_group<64><<<(R2 + 255) / 256, 256, 0, stream>>>(nx1, nx2, knn2, pool1, 32, S1P, S2P, bufA);

    // layer 0: A(pad64,Cin35) -> B
    k_zero<<<2, 256, 0, stream>>>(ssum, 512);
    k_gemm_wmma<64, 35, 64><<<(R2 / 16) / 8, 256, 0, stream>>>(bufA, W2[0], B2[0], bufB, R2);
    k_stats<<<256, 256, 0, stream>>>(bufB, S2P * KNN, 64, ssum, ssq);
    k_norm_relu<<<(int)(((size_t)R2 * 64 + 255) / 256), 256, 0, stream>>>(bufB, ssum, ssq, S2P * KNN, 64);
    // layer 1: B -> A
    k_zero<<<2, 256, 0, stream>>>(ssum, 512);
    k_gemm_wmma<64, 64, 64><<<(R2 / 16) / 8, 256, 0, stream>>>(bufB, W2[1], B2[1], bufA, R2);
    k_stats<<<256, 256, 0, stream>>>(bufA, S2P * KNN, 64, ssum, ssq);
    k_norm_relu<<<(int)(((size_t)R2 * 64 + 255) / 256), 256, 0, stream>>>(bufA, ssum, ssq, S2P * KNN, 64);
    // layer 2: A -> B
    k_zero<<<2, 256, 0, stream>>>(ssum, 512);
    k_gemm_wmma<64, 64, 64><<<(R2 / 16) / 8, 256, 0, stream>>>(bufA, W2[2], B2[2], bufB, R2);
    k_stats<<<256, 256, 0, stream>>>(bufB, S2P * KNN, 64, ssum, ssq);
    k_norm_relu<<<(int)(((size_t)R2 * 64 + 255) / 256), 256, 0, stream>>>(bufB, ssum, ssq, S2P * KNN, 64);

    // feat_l2 [B,64,S2] written transposed directly into d_out
    k_maxpool<<<(BATCH * S2P * 64 + 255) / 256, 256, 0, stream>>>(bufB, S2P, 64, out + FEAT_OFF, 1);

    // ---- outputs ----------------------------------------------------------
    k_out_pc<<<(BATCH * S1P * 3 + 255) / 256, 256, 0, stream>>>(nx1, S1P, out + PC1_OFF);
    k_out_pc<<<(BATCH * S2P * 3 + 255) / 256, 256, 0, stream>>>(nx2, S2P, out + PC2_OFF);
    k_copy_i32<<<(BATCH * S1P + 255) / 256, 256, 0, stream>>>(fps1, (int*)(out + FPS1_OFF), BATCH * S1P);
    k_copy_i32<<<(BATCH * S2P + 255) / 256, 256, 0, stream>>>(fps2, (int*)(out + FPS2_OFF), BATCH * S2P);
}
